// kMaxCrossAttentionLayer_83958020702412
// MI455X (gfx1250) — matrix operand order repaired and checked by reading, not compiled, not run on hardware
//
#include <hip/hip_runtime.h>

#define L 128
#define N 8
#define S 16384
#define C 256
#define Bc 256
#define V 256
#define MT 64          // s-rows per block in fused kernel

typedef __attribute__((ext_vector_type(16))) __bf16 v16bf;
typedef __attribute__((ext_vector_type(8)))  float  v8f;

union ABf { v16bf v; unsigned short h[16]; };

__device__ __forceinline__ unsigned short f2bf(float f) {
    union { float f; unsigned u; } x; x.f = f;
    unsigned r = x.u + 0x7FFFu + ((x.u >> 16) & 1u);   // round-to-nearest-even
    return (unsigned short)(r >> 16);
}

// monotonic unsigned ordering of float bits
__device__ __forceinline__ unsigned mono(float f) {
    unsigned u = __float_as_uint(f);
    return (u & 0x80000000u) ? ~u : (u | 0x80000000u);
}

__device__ __forceinline__ unsigned long long shflxor_u64(unsigned long long v, int msk) {
    unsigned lo = (unsigned)v, hi = (unsigned)(v >> 32);
    lo = __shfl_xor(lo, msk, 32);
    hi = __shfl_xor(hi, msk, 32);
    return ((unsigned long long)hi << 32) | lo;
}

// ---------------- kernel 0: Wp -> bf16 ----------------
__global__ __launch_bounds__(256) void k_convert_wp(const float* __restrict__ Wp,
                                                    unsigned short* __restrict__ wp_bf) {
    int i = blockIdx.x * 256 + threadIdx.x;   // 65536 elems
    wp_bf[i] = f2bf(Wp[i]);
}

// ---------------- kernel 1: q_proj (bf16, [N][L][B]) ----------------
__global__ __launch_bounds__(256) void k_qproj(const float* __restrict__ tgt,
                                               const float* __restrict__ qpos,
                                               const float* __restrict__ Wq,
                                               const float* __restrict__ bq,
                                               unsigned short* __restrict__ qbf) {
    __shared__ float q[C];
    int row = blockIdx.x;               // row = l*N + n
    int l = row >> 3, n = row & 7;
    int tid = threadIdx.x;
    q[tid] = tgt[(size_t)row * C + tid] + qpos[(size_t)row * C + tid];
    __syncthreads();
    float acc = bq[tid];
    const float* wrow = Wq + (size_t)tid * C;
    #pragma unroll 4
    for (int c = 0; c < C; ++c) acc += q[c] * wrow[c];
    qbf[((size_t)n * L + l) * Bc + tid] = f2bf(acc);
}

// ---------------- kernel 2: fused k_proj + normalize + logits + argmax ----------------
__global__ __launch_bounds__(256) void k_fused(const float* __restrict__ mem,
                                               const float* __restrict__ pos,
                                               const unsigned short* __restrict__ wp_bf,
                                               const float* __restrict__ bp,
                                               const unsigned short* __restrict__ qbf,
                                               unsigned long long* __restrict__ accum) {
    __shared__ unsigned short lds_k[MT][C + 8];   // k tile bf16; reused for k_norm
    __shared__ float          lds_kp[MT][C + 4];  // k_proj f32
    __shared__ float          lds_ss[MT][4];
    __shared__ float          lds_scale[MT];

    int blk = blockIdx.x;                 // N * S/MT blocks
    int n  = blk & 7;
    int s0 = (blk >> 3) * MT;
    int tid  = threadIdx.x;
    int wave = tid >> 5, lane = tid & 31;
    int m     = lane & 15;
    int half  = lane >> 4;
    int kbA   = half ? 16 : 0;    // A-frag K base (contiguous k = 16*half + s)
    int kbB   = half ? 16 : 0;    // B-frag K base (contiguous k = 16*half + s)
    int mbase = half ? 8 : 0;     // C/D row base

    // load k = memory + pos tile into LDS as bf16 (coalesced)
    for (int i = tid; i < MT * C; i += 256) {
        int r = i >> 8, c = i & 255;
        size_t g = ((size_t)(s0 + r) * N + n) * C + c;
        lds_k[r][c] = f2bf(mem[g] + pos[g]);
    }
    __syncthreads();

    // ---- stage A: k_proj = k @ Wp^T + bp  (WMMA bf16) ----
    for (int job = wave; job < (MT / 16) * 16; job += 8) {
        int mt = job >> 4, jt = job & 15;
        v8f acc = {0.f, 0.f, 0.f, 0.f, 0.f, 0.f, 0.f, 0.f};
        #pragma unroll
        for (int ks = 0; ks < C / 32; ++ks) {
            int k0 = ks * 32;
            ABf a, b;
            const unsigned short* arow = &lds_k[mt * 16 + m][k0];
            #pragma unroll
            for (int j = 0; j < 8; ++j) {
                int kk = kbA + 2 * j;
                a.h[2 * j]     = arow[kk];
                a.h[2 * j + 1] = arow[kk + 1];
            }
            const unsigned short* brow = wp_bf + (size_t)(jt * 16 + m) * C + k0 + kbB;
            *(uint4*)&b.h[0] = *(const uint4*)brow;
            *(uint4*)&b.h[8] = *(const uint4*)(brow + 8);
            acc = __builtin_amdgcn_wmma_f32_16x16x32_bf16(false, a.v, false, b.v,
                                                          (short)0, acc, false, false);
        }
        float bias = bp[jt * 16 + m];
        #pragma unroll
        for (int r = 0; r < 8; ++r)
            lds_kp[mt * 16 + mbase + r][jt * 16 + m] = acc[r] + bias;
    }
    __syncthreads();

    // ---- stage B: row L2 norms -> k_norm (bf16, reuses lds_k) ----
    {
        int r = tid >> 2, q = tid & 3;
        float ss = 0.f;
        #pragma unroll 4
        for (int c = q * 64; c < q * 64 + 64; ++c) { float v = lds_kp[r][c]; ss += v * v; }
        lds_ss[r][q] = ss;
    }
    __syncthreads();
    if (tid < MT) {
        float nrm = sqrtf(lds_ss[tid][0] + lds_ss[tid][1] + lds_ss[tid][2] + lds_ss[tid][3]);
        lds_scale[tid] = 1.0f / fmaxf(nrm, 1e-12f);
    }
    __syncthreads();
    for (int i = tid; i < MT * C; i += 256) {
        int r = i >> 8, c = i & 255;
        lds_k[r][c] = f2bf(lds_kp[r][c] * lds_scale[r]);
    }
    __syncthreads();

    // ---- stage C: logits = q_proj @ k_norm^T, fold into argmax accumulators ----
    {
        int lt = wave;                               // 8 waves == 8 l-tiles
        unsigned long long best[8];
        #pragma unroll
        for (int r = 0; r < 8; ++r) best[r] = 0ull;

        for (int st = 0; st < MT / 16; ++st) {
            v8f acc = {0.f, 0.f, 0.f, 0.f, 0.f, 0.f, 0.f, 0.f};
            #pragma unroll
            for (int ks = 0; ks < Bc / 32; ++ks) {
                int k0 = ks * 32;
                ABf a, b;
                const unsigned short* arow =
                    qbf + ((size_t)n * L + lt * 16 + m) * Bc + k0;
                #pragma unroll
                for (int j = 0; j < 8; ++j) {
                    int kk = kbA + 2 * j;
                    a.h[2 * j]     = arow[kk];
                    a.h[2 * j + 1] = arow[kk + 1];
                }
                const unsigned short* brow = &lds_k[st * 16 + m][k0 + kbB];
                #pragma unroll
                for (int i = 0; i < 16; ++i) b.h[i] = brow[i];
                acc = __builtin_amdgcn_wmma_f32_16x16x32_bf16(false, a.v, false, b.v,
                                                              (short)0, acc, false, false);
            }
            unsigned scol = (unsigned)(s0 + st * 16 + m);
            unsigned slow = 0xFFFFFFFFu - scol;      // smaller s wins ties under max
            #pragma unroll
            for (int r = 0; r < 8; ++r) {
                unsigned long long pk =
                    ((unsigned long long)mono(acc[r]) << 32) | slow;
                if (pk > best[r]) best[r] = pk;
            }
        }
        #pragma unroll
        for (int r = 0; r < 8; ++r) {
            unsigned long long v = best[r];
            #pragma unroll
            for (int msk = 1; msk < 16; msk <<= 1) {
                unsigned long long o = shflxor_u64(v, msk);
                if (o > v) v = o;
            }
            if (m == 0)
                atomicMax(&accum[(size_t)n * L + lt * 16 + mbase + r], v);
        }
    }
}

// ---------------- kernel 3: gather-recompute v, output proj, residual + LayerNorm ----------------
__global__ __launch_bounds__(256) void k_final(const unsigned long long* __restrict__ accum,
                                               const float* __restrict__ mem,
                                               const float* __restrict__ pos,
                                               const float* __restrict__ tgt,
                                               const float* __restrict__ Wp,
                                               const float* __restrict__ bp,
                                               const float* __restrict__ Wv,
                                               const float* __restrict__ bv,
                                               const float* __restrict__ Wo,
                                               const float* __restrict__ bo,
                                               const float* __restrict__ gamma,
                                               const float* __restrict__ beta,
                                               float* __restrict__ out) {
    __shared__ float v0[C];
    __shared__ float v1[C];
    __shared__ float red[256];
    int row = blockIdx.x;                 // row = l*N + n
    int l = row >> 3, n = row & 7;
    int tid = threadIdx.x;

    unsigned long long pk = accum[(size_t)n * L + l];
    int s = (int)(0xFFFFFFFFu - (unsigned)(pk & 0xFFFFFFFFull));

    // k row for winning s
    size_t g = ((size_t)s * N + n) * C + tid;
    v0[tid] = mem[g] + pos[g];
    __syncthreads();
    // k_proj row (f32)
    {
        float acc = bp[tid];
        const float* w = Wp + (size_t)tid * C;
        #pragma unroll 4
        for (int c = 0; c < C; ++c) acc += v0[c] * w[c];
        v1[tid] = acc;
    }
    __syncthreads();
    // v_proj row
    {
        float acc = bv[tid];
        const float* w = Wv + (size_t)tid * Bc;
        #pragma unroll 4
        for (int c = 0; c < Bc; ++c) acc += v1[c] * w[c];
        v0[tid] = acc;
    }
    __syncthreads();
    // output projection + residual
    float x;
    {
        float acc = bo[tid];
        const float* w = Wo + (size_t)tid * V;
        #pragma unroll 4
        for (int c = 0; c < V; ++c) acc += v0[c] * w[c];
        x = tgt[(size_t)row * C + tid] + acc;
    }
    // LayerNorm over C
    red[tid] = x;
    __syncthreads();
    #pragma unroll
    for (int off = 128; off > 0; off >>= 1) {
        if (tid < off) red[tid] += red[tid + off];
        __syncthreads();
    }
    float mu = red[0] * (1.0f / C);
    __syncthreads();
    float d = x - mu;
    red[tid] = d * d;
    __syncthreads();
    #pragma unroll
    for (int off = 128; off > 0; off >>= 1) {
        if (tid < off) red[tid] += red[tid + off];
        __syncthreads();
    }
    float var = red[0] * (1.0f / C);
    out[(size_t)row * C + tid] = d * rsqrtf(var + 1e-5f) * gamma[tid] + beta[tid];
}

extern "C" void kernel_launch(void* const* d_in, const int* in_sizes, int n_in,
                              void* d_out, int out_size, void* d_ws, size_t ws_size,
                              hipStream_t stream) {
    const float* tgt    = (const float*)d_in[0];
    const float* memory = (const float*)d_in[1];
    const float* pos    = (const float*)d_in[2];
    const float* qpos   = (const float*)d_in[3];
    const float* Wq = (const float*)d_in[4];  const float* bq = (const float*)d_in[5];
    const float* Wp = (const float*)d_in[6];  const float* bp = (const float*)d_in[7];
    const float* Wv = (const float*)d_in[8];  const float* bv = (const float*)d_in[9];
    const float* Wo = (const float*)d_in[10]; const float* bo = (const float*)d_in[11];
    const float* gamma = (const float*)d_in[12];
    const float* beta  = (const float*)d_in[13];
    float* out = (float*)d_out;

    // workspace layout
    unsigned short* wp_bf = (unsigned short*)d_ws;                       // 256*256*2   = 131072 B
    unsigned short* qbf   = (unsigned short*)((char*)d_ws + 131072);     // 8*128*256*2 = 524288 B
    unsigned long long* accum =
        (unsigned long long*)((char*)d_ws + 131072 + 524288);            // 1024*8      = 8192 B

    hipMemsetAsync(accum, 0, (size_t)N * L * sizeof(unsigned long long), stream);

    k_convert_wp<<<Bc * C / 256, 256, 0, stream>>>(Wp, wp_bf);
    k_qproj<<<L * N, 256, 0, stream>>>(tgt, qpos, Wq, bq, qbf);
    k_fused<<<N * (S / MT), 256, 0, stream>>>(memory, pos, wp_bf, bp, qbf, accum);
    k_final<<<L * N, 256, 0, stream>>>(accum, memory, pos, tgt, Wp, bp, Wv, bv,
                                       Wo, bo, gamma, beta, out);
}
